// Linear4Bit_51101520888045
// MI455X (gfx1250) — matrix operand order repaired and checked by reading, not compiled, not run on hardware
//
#include <hip/hip_runtime.h>

// ---- WMMA vector types (wave32, gfx1250) ----
typedef _Float16 v16h __attribute__((ext_vector_type(16)));
typedef _Float16 v8h  __attribute__((ext_vector_type(8)));
typedef _Float16 v4h  __attribute__((ext_vector_type(4)));
typedef _Float16 v2h  __attribute__((ext_vector_type(2)));
typedef float    v8f  __attribute__((ext_vector_type(8)));

#define O_DIM  11008
#define I_DIM  4096
#define G_SZ   16
#define B_TOK  64
#define GPR    (I_DIM / G_SZ)   // 256 groups per output row
#define KSTEPS (I_DIM / 32)     // 128 WMMA K-steps
#define KCHUNK 4                // K-split factor (occupancy)
#define OUT_N  (B_TOK * O_DIM)  // 704512

// ---------------------------------------------------------------------------
// Kernel 1: convert x [64,4096] f32 -> f16 into workspace (row-major).
// ---------------------------------------------------------------------------
__global__ __launch_bounds__(256) void xcvt_kernel(const float* __restrict__ x,
                                                   _Float16* __restrict__ xh,
                                                   int n) {
  int idx = (blockIdx.x * blockDim.x + threadIdx.x) * 4;
  if (idx + 3 < n) {
    float4 v = *(const float4*)(x + idx);
    v4h h = { (_Float16)v.x, (_Float16)v.y, (_Float16)v.z, (_Float16)v.w };
    *(v4h*)(xh + idx) = h;
  }
}

// ---------------------------------------------------------------------------
// Per-K-step register fragment set (for software pipelining).
// ---------------------------------------------------------------------------
struct Frag {
  int4     q0, q1;   // 8 int32 = one quant group (16 codes)
  _Float16 nrm;      // group norm
  v16h     a[4];     // A fragments for 4 M-tiles
};

__device__ __forceinline__ void load_step(int ks, int gbase, int hi, int nlane,
                                          const int* __restrict__ wq,
                                          const _Float16* __restrict__ wn,
                                          const _Float16* __restrict__ xh,
                                          Frag& f) {
  const int g = gbase + ks * 2 + hi;
  const int4* qp = (const int4*)(wq + g * 8);
  __builtin_prefetch((const void*)((const char*)qp + 1024), 0, 1);  // 16 steps ahead
  f.q0 = qp[0];
  f.q1 = qp[1];
  f.nrm = wn[g];
  const int kBase = ks * 32;
#pragma unroll
  for (int mt = 0; mt < 4; ++mt) {
    const _Float16* xr = xh + (mt * 16 + nlane) * I_DIM + kBase + hi * 16;
    v8h alo = *(const v8h*)(xr);
    v8h ahi = *(const v8h*)(xr + 8);
#pragma unroll
    for (int i = 0; i < 8; ++i) { f.a[mt][i] = alo[i]; f.a[mt][8 + i] = ahi[i]; }
  }
}

__device__ __forceinline__ void compute_step(const Frag& f,
                                             const unsigned int* lut,
                                             v8f acc[4]) {
  v2h n2 = { f.nrm, f.nrm };
  // Code bytes are 0..255 by construction; LDS OOB reads return 0 (no fault),
  // so no defensive mask -> 1 shift + 1 ds_load + 1 v_pk_mul per byte.
  unsigned int qv[8] = { (unsigned)f.q0.x, (unsigned)f.q0.y, (unsigned)f.q0.z, (unsigned)f.q0.w,
                         (unsigned)f.q1.x, (unsigned)f.q1.y, (unsigned)f.q1.z, (unsigned)f.q1.w };
  v16h bfrag;
#pragma unroll
  for (int e = 0; e < 8; ++e) {
    union { unsigned int u; v2h h; } t;
    t.u = lut[qv[e]];            // ds_load_b32
    t.h = t.h * n2;              // v_pk_mul_f16
    bfrag[2 * e]     = t.h[0];
    bfrag[2 * e + 1] = t.h[1];
  }
#pragma unroll
  for (int mt = 0; mt < 4; ++mt) {
    acc[mt] = __builtin_amdgcn_wmma_f32_16x16x32_f16(
        false, f.a[mt], false, bfrag, (short)0, acc[mt], false, false);
  }
}

// ---------------------------------------------------------------------------
// Kernel 2: 4-bit dequant + WMMA GEMM, software-pipelined.
//   grid = (O/64, kchunks), 128 threads (4 waves).
//   Wave owns 16 output features x all 64 tokens over a K range.
// ---------------------------------------------------------------------------
__global__ __launch_bounds__(128) void gemm4bit_kernel(
    const _Float16* __restrict__ xh,
    const int*      __restrict__ wq,
    const _Float16* __restrict__ wn,
    const float*    __restrict__ bias,
    float*          __restrict__ dst_base,
    int steps_per_chunk, int add_bias) {
  __shared__ unsigned int lut[256];
  int tid = threadIdx.x;
  for (int e = tid; e < 256; e += 128) {
    union { unsigned int u; v2h h; } t;
    t.h[0] = (_Float16)((float)(e & 15) * (2.0f / 15.0f) - 1.0f);
    t.h[1] = (_Float16)((float)(e >> 4) * (2.0f / 15.0f) - 1.0f);
    lut[e] = t.u;
  }
  __syncthreads();

  const int wave  = tid >> 5;
  const int lane  = tid & 31;
  const int nlane = lane & 15;
  const int hi    = lane >> 4;
  const int o     = blockIdx.x * 64 + wave * 16 + nlane;
  const int gbase = o * GPR;
  const int k0    = blockIdx.y * steps_per_chunk;
  float* dst      = dst_base + blockIdx.y * OUT_N;

  v8f acc[4] = {v8f{}, v8f{}, v8f{}, v8f{}};

  // Software pipeline, 2x unrolled ping-pong (steps_per_chunk is even).
  Frag f0, f1;
  load_step(k0, gbase, hi, nlane, wq, wn, xh, f0);
  int ks = k0;
  for (int it = 0; it < steps_per_chunk; it += 2) {
    load_step(ks + 1, gbase, hi, nlane, wq, wn, xh, f1);   // issue loads
    compute_step(f0, lut, acc);                             // consume prev
    int ks2 = (it + 2 < steps_per_chunk) ? ks + 2 : ks + 1; // clamp last
    load_step(ks2, gbase, hi, nlane, wq, wn, xh, f0);
    compute_step(f1, lut, acc);
    ks += 2;
  }

  // Epilogue: C/D layout — VGPR r: M = mt*16 + hi*8 + r, N = nlane.
  const float bo = add_bias ? bias[o] : 0.0f;
#pragma unroll
  for (int mt = 0; mt < 4; ++mt) {
#pragma unroll
    for (int r = 0; r < 8; ++r) {
      int m = mt * 16 + hi * 8 + r;
      dst[m * O_DIM + o] = acc[mt][r] + bo;
    }
  }
}

// ---------------------------------------------------------------------------
// Kernel 3: deterministic reduction of KCHUNK partials + bias -> out.
// ---------------------------------------------------------------------------
__global__ __launch_bounds__(256) void reduce_kernel(const float* __restrict__ p,
                                                     const float* __restrict__ bias,
                                                     float* __restrict__ out) {
  int idx = (blockIdx.x * blockDim.x + threadIdx.x) * 4;
  if (idx + 3 < OUT_N) {
    float4 s = *(const float4*)(p + idx);
#pragma unroll
    for (int c = 1; c < KCHUNK; ++c) {
      float4 q = *(const float4*)(p + c * OUT_N + idx);
      s.x += q.x; s.y += q.y; s.z += q.z; s.w += q.w;
    }
    int oo = idx % O_DIM;                       // O_DIM % 4 == 0 -> same row
    float4 b = *(const float4*)(bias + oo);
    s.x += b.x; s.y += b.y; s.z += b.z; s.w += b.w;
    *(float4*)(out + idx) = s;
  }
}

// ---------------------------------------------------------------------------
// Launch: inputs: x[f32], weight_q4[i32], weight_norm[f16], bias[f32].
// d_ws: [0,512KB) f16 x; [512KB, +KCHUNK*2.82MB) K-split partials.
// ---------------------------------------------------------------------------
extern "C" void kernel_launch(void* const* d_in, const int* in_sizes, int n_in,
                              void* d_out, int out_size, void* d_ws, size_t ws_size,
                              hipStream_t stream) {
  const float*    x    = (const float*)d_in[0];
  const int*      wq   = (const int*)d_in[1];
  const _Float16* wn   = (const _Float16*)d_in[2];
  const float*    bias = (const float*)d_in[3];
  float*          out  = (float*)d_out;

  const size_t xh_bytes   = (size_t)B_TOK * I_DIM * sizeof(_Float16); // 512KB
  const size_t part_bytes = (size_t)OUT_N * sizeof(float);            // 2.82MB
  _Float16* xh = (_Float16*)d_ws;

  const int n = B_TOK * I_DIM;
  xcvt_kernel<<<n / (256 * 4), 256, 0, stream>>>(x, xh, n);

  if (ws_size >= xh_bytes + (size_t)KCHUNK * part_bytes) {
    // K-split path: 4x the waves for latency hiding, deterministic reduce.
    float* partial = (float*)((char*)d_ws + xh_bytes);
    dim3 grid(O_DIM / 64, KCHUNK);
    gemm4bit_kernel<<<grid, 128, 0, stream>>>(xh, wq, wn, bias, partial,
                                              KSTEPS / KCHUNK, /*add_bias=*/0);
    reduce_kernel<<<OUT_N / (256 * 4), 256, 0, stream>>>(partial, bias, out);
  } else {
    // Fallback: single K pass straight to out (still pipelined).
    dim3 grid(O_DIM / 64, 1);
    gemm4bit_kernel<<<grid, 128, 0, stream>>>(xh, wq, wn, bias, out,
                                              KSTEPS, /*add_bias=*/1);
  }
}
